// MultiHeadSelfAttention_13486197309752
// MI455X (gfx1250) — matrix operand order repaired and checked by reading, not compiled, not run on hardware
//
#include <hip/hip_runtime.h>
#include <hip/hip_bf16.h>

// MHA forward for B=2, S=2048, D=1024, H=16 (hd=64) on gfx1250 (wave32, WMMA).
// Pipeline: [proj QKV: f32 GEMM -> bf16 head-split] -> [flash attention bf16/f32]
//           -> [out projection bf16 x f32 W_T -> f32 out]
// Uses GLOBAL_LOAD_ASYNC_TO_LDS_B64 (ASYNCcnt) for bf16 tile staging.
// Workspace requirement: 4 x (2*2048*1024) bf16 = 32 MB.

#define BDIM   2
#define SLEN   2048
#define DMODEL 1024
#define NHEAD  16
#define HD     64

typedef __attribute__((ext_vector_type(16))) __bf16 v16bf;
typedef __attribute__((ext_vector_type(8)))  float  v8f;

union FragU { v16bf v; unsigned u[8]; };

__device__ __forceinline__ unsigned short f32_to_bf16(float f) {
  union { float f; unsigned u; } x; x.f = f;
  unsigned r = x.u + 0x7fffu + ((x.u >> 16) & 1u);   // round-to-nearest-even
  return (unsigned short)(r >> 16);
}
__device__ __forceinline__ unsigned pack_bf16x2(float lo, float hi) {
  return (unsigned)f32_to_bf16(lo) | ((unsigned)f32_to_bf16(hi) << 16);
}

__device__ __forceinline__ v8f wmma_bf16(v16bf a, v16bf b, v8f c) {
  return __builtin_amdgcn_wmma_f32_16x16x32_bf16(false, a, false, b, (short)0, c,
                                                 false, false);
}

// Low 32 bits of a generic pointer to LDS = LDS byte offset (flat aperture rule).
__device__ __forceinline__ unsigned lds_off(const void* p) {
  return (unsigned)(unsigned long long)p;
}

// Async copy: 8 bytes global -> LDS, per lane (GV mode). Tracked by ASYNCcnt.
__device__ __forceinline__ void async_b64(unsigned lds_addr, const void* gptr) {
  unsigned long long ga = (unsigned long long)gptr;
  asm volatile("global_load_async_to_lds_b64 %0, %1, off"
               :: "v"(lds_addr), "v"(ga) : "memory");
}
__device__ __forceinline__ void wait_async0() {
  asm volatile("s_wait_asynccnt 0x0" ::: "memory");
}

// A fragment (16x32, bf16) from row-major LDS [row][k], stride strideH (bf16 units).
// Layout: lane m=ln; vgpr i: K=half*16+2i (contiguous K order, matches B frags).
__device__ __forceinline__ v16bf load_a_frag(const unsigned short* base, int row0,
                                             int strideH, int kOff, int half, int ln) {
  FragU f;
  const unsigned short* p = base + (size_t)(row0 + ln) * strideH + kOff + half * 16;
#pragma unroll
  for (int i = 0; i < 8; ++i) {
    f.u[i]     = *(const unsigned*)(p + 2 * i);
  }
  return f.v;
}

// B fragment (32x16, bf16) from pair-packed LDS: lB[kp][n] = (B[2kp][n], B[2kp+1][n]).
// Layout: lanes 0-15 K=0..15 (kp=j), lanes 16-31 K=16..31 (kp=8+j); n = ln.
__device__ __forceinline__ v16bf load_b_frag(const unsigned* lB, int kpBase,
                                             int strideW, int nOff, int half, int ln) {
  FragU f;
  const unsigned* p = lB + (size_t)(kpBase + half * 8) * strideW + nOff + ln;
#pragma unroll
  for (int j = 0; j < 8; ++j) f.u[j] = p[j * strideW];
  return f.v;
}

// B fragment from u64-granule K layout: lK64[c][key] = 4 consecutive d of one key,
// i.e. pairs (2c, 2c+1). u32 index = ((dp>>1)*68 + key)*2 + (dp&1).
__device__ __forceinline__ v16bf load_kb_frag(const unsigned* lK, int kpBase,
                                              int nOff, int half, int ln) {
  FragU f;
#pragma unroll
  for (int j = 0; j < 8; ++j) {
    int dp = kpBase + half * 8 + j;
    f.u[j] = lK[(((dp >> 1) * 68) + nOff + ln) * 2 + (dp & 1)];
  }
  return f.v;
}

// ---------------------------------------------------------------------------
// Kernel 1: fused QKV projection.  C[m,n] = A[m,:] @ W[:,n], M=B*S=4096,
// K=N=1024.  Block tile 128x128, 8 waves, each wave a 32x64 subtile.
// Output bf16, head-split layout [B][H][S][64].
// ---------------------------------------------------------------------------
__global__ __launch_bounds__(256, 1) void proj_qkv_kernel(
    const float* __restrict__ qin, const float* __restrict__ kin,
    const float* __restrict__ vin, const float* __restrict__ Wq,
    const float* __restrict__ Wk, const float* __restrict__ Wv,
    unsigned short* __restrict__ Qo, unsigned short* __restrict__ Ko,
    unsigned short* __restrict__ Vo) {
  __shared__ __attribute__((aligned(16))) unsigned short lA[128 * 36];
  __shared__ __attribute__((aligned(16))) unsigned       lB[16 * 132];

  const int zz = blockIdx.z;
  const float* A  = (zz == 0) ? qin : (zz == 1) ? kin : vin;
  const float* Bw = (zz == 0) ? Wq  : (zz == 1) ? Wk  : Wv;
  unsigned short* O = (zz == 0) ? Qo : (zz == 1) ? Ko : Vo;

  const int tid = threadIdx.x;
  const int wid = tid >> 5, lane = tid & 31, half = lane >> 4, ln = lane & 15;
  const int waveM = wid >> 1, waveN = wid & 1;
  const int mBase = blockIdx.y * 128, nBase = blockIdx.x * 128;

  v8f acc[2][4] = {};

  for (int kk = 0; kk < DMODEL; kk += 32) {
    __syncthreads();
    // A tile 128x32 f32 -> bf16 LDS (row-major, stride 36)
#pragma unroll
    for (int i = 0; i < 4; ++i) {
      int lin = tid + i * 256;
      int row = lin >> 3, c4 = lin & 7;
      const float4 a4 =
          *(const float4*)(A + (size_t)(mBase + row) * DMODEL + kk + c4 * 4);
      unsigned* dst = (unsigned*)&lA[row * 36 + c4 * 4];
      dst[0] = pack_bf16x2(a4.x, a4.y);
      dst[1] = pack_bf16x2(a4.z, a4.w);
    }
    // B tile 32x128 f32 -> K-pair-packed bf16 LDS (stride 132 u32)
#pragma unroll
    for (int i = 0; i < 2; ++i) {
      int lin = tid + i * 256;
      int kp = lin >> 5, n4 = lin & 31;
      const float4 x =
          *(const float4*)(Bw + (size_t)(kk + 2 * kp) * DMODEL + nBase + n4 * 4);
      const float4 y =
          *(const float4*)(Bw + (size_t)(kk + 2 * kp + 1) * DMODEL + nBase + n4 * 4);
      uint4 o;
      o.x = pack_bf16x2(x.x, y.x);
      o.y = pack_bf16x2(x.y, y.y);
      o.z = pack_bf16x2(x.z, y.z);
      o.w = pack_bf16x2(x.w, y.w);
      *(uint4*)&lB[kp * 132 + n4 * 4] = o;
    }
    if (kk + 32 < DMODEL)
      __builtin_prefetch(A + (size_t)(mBase + (tid >> 1)) * DMODEL + kk + 32, 0, 1);
    __syncthreads();

    v16bf af[2], bfr[4];
#pragma unroll
    for (int mt = 0; mt < 2; ++mt)
      af[mt] = load_a_frag(lA, waveM * 32 + mt * 16, 36, 0, half, ln);
#pragma unroll
    for (int nt = 0; nt < 4; ++nt)
      bfr[nt] = load_b_frag(lB, 0, 132, waveN * 64 + nt * 16, half, ln);
#pragma unroll
    for (int mt = 0; mt < 2; ++mt)
#pragma unroll
      for (int nt = 0; nt < 4; ++nt)
        acc[mt][nt] = wmma_bf16(af[mt], bfr[nt], acc[mt][nt]);
  }

  // Store bf16, head-split [B][H][S][64]
#pragma unroll
  for (int mt = 0; mt < 2; ++mt)
#pragma unroll
    for (int nt = 0; nt < 4; ++nt)
#pragma unroll
      for (int r = 0; r < 8; ++r) {
        int m = mBase + waveM * 32 + mt * 16 + half * 8 + r;
        int n = nBase + waveN * 64 + nt * 16 + ln;
        int b = m >> 11, s = m & (SLEN - 1);
        int h = n >> 6, d = n & 63;
        O[(((size_t)b * NHEAD + h) * SLEN + s) * HD + d] =
            f32_to_bf16(acc[mt][nt][r]);
      }
}

// ---------------------------------------------------------------------------
// Kernel 2: flash attention.  Block = 64 q-rows of one (b,h); 4 waves, each
// owning 16 q-rows.  Q/K tiles staged by async global->LDS copies.
// ---------------------------------------------------------------------------
__global__ __launch_bounds__(128, 1) void attn_kernel(
    const unsigned short* __restrict__ Qw, const unsigned short* __restrict__ Kw,
    const unsigned short* __restrict__ Vw, const int* __restrict__ mask,
    unsigned short* __restrict__ Ctx) {
  __shared__ __attribute__((aligned(16))) unsigned short lQ[64 * 68];
  __shared__ __attribute__((aligned(16))) unsigned       lK[16 * 68 * 2];
  __shared__ __attribute__((aligned(16))) unsigned       lV[32 * 68];
  __shared__ __attribute__((aligned(16))) unsigned short lP[64 * 68];

  const int tid = threadIdx.x, wid = tid >> 5, lane = tid & 31;
  const int half = lane >> 4, ln = lane & 15;
  const int qt = blockIdx.x, bh = blockIdx.y;
  const int b = bh >> 4, h = bh & 15;
  const int qbase = qt * 64;
  const unsigned short* Qh = Qw + (size_t)bh * SLEN * HD;
  const unsigned short* Kh = Kw + (size_t)bh * SLEN * HD;
  const unsigned short* Vh = Vw + (size_t)bh * SLEN * HD;
  const float LOG2E = 1.4426950408889634f;
  const unsigned lQ0 = lds_off(&lQ[0]);
  const unsigned lK0 = lds_off(&lK[0]);

  // Q tile 64x64 bf16, row-major stride 68: async b64 copies (8B-aligned rows).
#pragma unroll
  for (int i = 0; i < 4; ++i) {
    int lin = tid + i * 128;                // 512 tasks
    int row = lin >> 3, c8 = lin & 7;       // 8 qwords per 64-elem row
    async_b64(lQ0 + row * 136 + c8 * 8,
              Qh + (size_t)(qbase + row) * HD + c8 * 4);
  }

  float mrow[8], lrow[8];
  v8f oacc[4] = {};
#pragma unroll
  for (int r = 0; r < 8; ++r) { mrow[r] = -3.0e38f; lrow[r] = 0.f; }

  for (int kb = 0; kb < SLEN; kb += 64) {
    __syncthreads();
    // K tile: async b64 restructure.  lK64[c][key] <- K[kb+key][4c..4c+3];
    // gives packed d-pairs (2c,2c+1) per u32.  c=0..15, key=0..63.
#pragma unroll
    for (int i = 0; i < 8; ++i) {
      int t = tid + i * 128;                // 1024 tasks
      int c = t >> 6, key = t & 63;
      async_b64(lK0 + (c * 68 + key) * 8,
                Kh + (size_t)(kb + key) * HD + c * 4);
    }
    // V tile (manual: interleaves two rows): lV[kp][d] = (V[2kp][d], V[2kp+1][d])
#pragma unroll
    for (int i = 0; i < 4; ++i) {
      int lin = tid + i * 128;
      int kp = lin >> 4, c4 = lin & 15;
      uint2 va = *(const uint2*)(Vh + (size_t)(kb + 2 * kp) * HD + c4 * 4);
      uint2 vb = *(const uint2*)(Vh + (size_t)(kb + 2 * kp + 1) * HD + c4 * 4);
      uint4 o;
      o.x = (va.x & 0xffffu) | (vb.x << 16);
      o.y = (va.x >> 16)     | (vb.x & 0xffff0000u);
      o.z = (va.y & 0xffffu) | (vb.y << 16);
      o.w = (va.y >> 16)     | (vb.y & 0xffff0000u);
      *(uint4*)&lV[kp * 68 + c4 * 4] = o;
    }
    wait_async0();
    __syncthreads();

    // Scores: S = Q x K^T  (16 q-rows x 64 keys per wave)
    v8f sacc[4] = {};
#pragma unroll
    for (int ks = 0; ks < 2; ++ks) {
      v16bf qa = load_a_frag(lQ, wid * 16, 68, ks * 32, half, ln);
#pragma unroll
      for (int nt = 0; nt < 4; ++nt) {
        v16bf kbf = load_kb_frag(lK, ks * 16, nt * 16, half, ln);
        sacc[nt] = wmma_bf16(qa, kbf, sacc[nt]);
      }
    }

    // scale + mask
    float st[4][8];
#pragma unroll
    for (int nt = 0; nt < 4; ++nt)
#pragma unroll
      for (int r = 0; r < 8; ++r) {
        float sv = sacc[nt][r] * 0.125f;  // 1/sqrt(64)
        int qrow = qbase + wid * 16 + half * 8 + r;
        int kcol = kb + nt * 16 + ln;
        if (mask[(size_t)qrow * SLEN + kcol] == 0) sv = -1.0e9f;
        st[nt][r] = sv;
      }

    // online softmax (row stats per C-layout row; reduce across 16-lane group)
#pragma unroll
    for (int r = 0; r < 8; ++r) {
      float tmax = fmaxf(fmaxf(st[0][r], st[1][r]), fmaxf(st[2][r], st[3][r]));
#pragma unroll
      for (int off = 8; off >= 1; off >>= 1)
        tmax = fmaxf(tmax, __shfl_xor(tmax, off, 32));
      float mnew = fmaxf(mrow[r], tmax);
      float alpha = __builtin_exp2f((mrow[r] - mnew) * LOG2E);
      mrow[r] = mnew;
      float rs = 0.f;
#pragma unroll
      for (int nt = 0; nt < 4; ++nt) {
        float pv = __builtin_exp2f((st[nt][r] - mnew) * LOG2E);
        st[nt][r] = pv;
        rs += pv;
      }
#pragma unroll
      for (int off = 8; off >= 1; off >>= 1) rs += __shfl_xor(rs, off, 32);
      lrow[r] = lrow[r] * alpha + rs;
#pragma unroll
      for (int nt = 0; nt < 4; ++nt) oacc[nt][r] *= alpha;
      int prow = wid * 16 + half * 8 + r;
#pragma unroll
      for (int nt = 0; nt < 4; ++nt)
        lP[prow * 68 + nt * 16 + ln] = f32_to_bf16(st[nt][r]);
    }

    // O += P x V  (contract over 64 keys; per-wave P rows, no cross-wave sync)
#pragma unroll
    for (int ks = 0; ks < 2; ++ks) {
      v16bf pa = load_a_frag(lP, wid * 16, 68, ks * 32, half, ln);
#pragma unroll
      for (int nt = 0; nt < 4; ++nt) {
        v16bf vbf = load_b_frag(lV, ks * 16, 68, nt * 16, half, ln);
        oacc[nt] = wmma_bf16(pa, vbf, oacc[nt]);
      }
    }
  }

  // epilogue: O/l, merge heads -> Ctx [B][S][D] bf16
#pragma unroll
  for (int nt = 0; nt < 4; ++nt)
#pragma unroll
    for (int r = 0; r < 8; ++r) {
      float o = oacc[nt][r] / lrow[r];
      int srow = qbase + wid * 16 + half * 8 + r;
      int d = nt * 16 + ln;
      Ctx[((size_t)b * SLEN + srow) * DMODEL + h * HD + d] = f32_to_bf16(o);
    }
}

// ---------------------------------------------------------------------------
// Kernel 3: output projection.  out[m,n] = Ctx[m,:] @ W_T[:,n], f32 out.
// A tile (bf16) staged with async global->LDS copies.
// ---------------------------------------------------------------------------
__global__ __launch_bounds__(256, 1) void out_proj_kernel(
    const unsigned short* __restrict__ Ctx, const float* __restrict__ Wt,
    float* __restrict__ Out) {
  __shared__ __attribute__((aligned(16))) unsigned short lA[128 * 36];
  __shared__ __attribute__((aligned(16))) unsigned       lB[16 * 132];

  const int tid = threadIdx.x;
  const int wid = tid >> 5, lane = tid & 31, half = lane >> 4, ln = lane & 15;
  const int waveM = wid >> 1, waveN = wid & 1;
  const int mBase = blockIdx.y * 128, nBase = blockIdx.x * 128;
  const unsigned lA0 = lds_off(&lA[0]);

  v8f acc[2][4] = {};

  for (int kk = 0; kk < DMODEL; kk += 32) {
    __syncthreads();
    // A tile 128x32 bf16: async b64 copies (rows 72B, 8B-aligned chunks)
#pragma unroll
    for (int i = 0; i < 4; ++i) {
      int lin = tid + i * 256;              // 1024 tasks
      int row = lin >> 3, c8 = lin & 7;
      async_b64(lA0 + row * 72 + c8 * 8,
                Ctx + (size_t)(mBase + row) * DMODEL + kk + c8 * 4);
    }
    // B tile 32x128 f32 -> K-pair-packed bf16 LDS
#pragma unroll
    for (int i = 0; i < 2; ++i) {
      int lin = tid + i * 256;
      int kp = lin >> 5, n4 = lin & 31;
      const float4 x =
          *(const float4*)(Wt + (size_t)(kk + 2 * kp) * DMODEL + nBase + n4 * 4);
      const float4 y =
          *(const float4*)(Wt + (size_t)(kk + 2 * kp + 1) * DMODEL + nBase + n4 * 4);
      uint4 o;
      o.x = pack_bf16x2(x.x, y.x);
      o.y = pack_bf16x2(x.y, y.y);
      o.z = pack_bf16x2(x.z, y.z);
      o.w = pack_bf16x2(x.w, y.w);
      *(uint4*)&lB[kp * 132 + n4 * 4] = o;
    }
    wait_async0();
    __syncthreads();

    v16bf af[2], bfr[4];
#pragma unroll
    for (int mt = 0; mt < 2; ++mt)
      af[mt] = load_a_frag(lA, waveM * 32 + mt * 16, 36, 0, half, ln);
#pragma unroll
    for (int nt = 0; nt < 4; ++nt)
      bfr[nt] = load_b_frag(lB, 0, 132, waveN * 64 + nt * 16, half, ln);
#pragma unroll
    for (int mt = 0; mt < 2; ++mt)
#pragma unroll
      for (int nt = 0; nt < 4; ++nt)
        acc[mt][nt] = wmma_bf16(af[mt], bfr[nt], acc[mt][nt]);
  }

#pragma unroll
  for (int mt = 0; mt < 2; ++mt)
#pragma unroll
    for (int nt = 0; nt < 4; ++nt)
#pragma unroll
      for (int r = 0; r < 8; ++r) {
        int m = mBase + waveM * 32 + mt * 16 + half * 8 + r;
        int n = nBase + waveN * 64 + nt * 16 + ln;
        Out[(size_t)m * DMODEL + n] = acc[mt][nt][r];
      }
}

// ---------------------------------------------------------------------------
extern "C" void kernel_launch(void* const* d_in, const int* in_sizes, int n_in,
                              void* d_out, int out_size, void* d_ws, size_t ws_size,
                              hipStream_t stream) {
  const float* q    = (const float*)d_in[0];
  const float* k    = (const float*)d_in[1];
  const float* v    = (const float*)d_in[2];
  const int*   mask = (const int*)d_in[3];
  const float* Wq   = (const float*)d_in[4];
  const float* Wk   = (const float*)d_in[5];
  const float* Wv   = (const float*)d_in[6];
  const float* Wt   = (const float*)d_in[7];
  float* out = (float*)d_out;

  const size_t NE = (size_t)BDIM * SLEN * DMODEL;  // 4M elements
  unsigned short* Qw = (unsigned short*)d_ws;
  unsigned short* Kw = Qw + NE;
  unsigned short* Vw = Kw + NE;
  unsigned short* Cw = Vw + NE;

  dim3 gp(DMODEL / 128, (BDIM * SLEN) / 128, 3);   // 8 x 32 x 3
  proj_qkv_kernel<<<gp, 256, 0, stream>>>(q, k, v, Wq, Wk, Wv, Qw, Kw, Vw);

  dim3 ga(SLEN / 64, BDIM * NHEAD);                // 32 x 32
  attn_kernel<<<ga, 128, 0, stream>>>(Qw, Kw, Vw, mask, Cw);

  dim3 gf(DMODEL / 128, (BDIM * SLEN) / 128);      // 8 x 32
  out_proj_kernel<<<gf, 256, 0, stream>>>(Cw, Wt, out);
}
